// MultiQuerySelfAttention_14955076124979
// MI455X (gfx1250) — matrix operand order repaired and checked by reading, not compiled, not run on hardware
//
#include <hip/hip_runtime.h>
#include <stdint.h>

// ---------------------------------------------------------------------------
// MQA self-attention, CDNA5 (gfx1250) WMMA bf16 pipeline with async-LDS
// double-buffered staging (GLOBAL_LOAD_ASYNC_TO_LDS_B128 / ASYNCcnt).
//   x:(4,2048,2048) f32 -> bf16
//   Q = x@qwT (8192x2048), K = x@kwT (8192x128), V^T = (x@vwT)^T  [b][e][t]
//   flash-style attention, faithful -1e-9 mask fill (analytic, mask not read)
//   attention output stored transposed [b][h][e][t]  (reference's
//   swapaxes(-1,-2).reshape)
//   out = attnT @ owT + ob   (f32 result)
// ---------------------------------------------------------------------------

typedef __attribute__((ext_vector_type(16))) __bf16 v16bf;
typedef __attribute__((ext_vector_type(8)))  __bf16 v8bf;
typedef __attribute__((ext_vector_type(8)))  float  v8f;

#define DIM 2048
#define SEQ 2048
#define BS  4
#define NH  16
#define HD  128
#define MTOT (BS * SEQ)   // 8192

// padded LDS row strides (elements): 36 / 68 dwords -> conflict-free
#define PAD      8
#define AROW     (64 + PAD)     // 72 elems = 144B
#define KROW     (128 + PAD)    // 136 elems = 272B

// ---------------------------------------------------------------------------
// helpers
// ---------------------------------------------------------------------------
__device__ __forceinline__ v16bf combine8(v8bf lo, v8bf hi) {
  v16bf r;
#pragma unroll
  for (int i = 0; i < 8; ++i) { r[i] = lo[i]; r[i + 8] = hi[i]; }
  return r;
}

// A-matrix 16x32 bf16 fragment (this code's contiguous k order, matched by B):
//   lanes 0-15 (row m=lane): K = 0..15; lanes 16-31 (row m=lane-16):
//   K = 16..31.  rowbase points at element K=0 of this lane's row.
__device__ __forceinline__ v16bf load_a_frag(const __bf16* rowbase, int half) {
  const __bf16* p = rowbase + half * 16;
  v8bf lo = *(const v8bf*)(p);
  v8bf hi = *(const v8bf*)(p + 8);
  return combine8(lo, hi);
}

// B-matrix 32x16 fragment: lane (ln,half) holds column n, K = half*16 + 0..15
__device__ __forceinline__ v16bf load_b_frag(const __bf16* p) {
  v8bf lo = *(const v8bf*)(p);
  v8bf hi = *(const v8bf*)(p + 8);
  return combine8(lo, hi);
}

__device__ __forceinline__ v8f wmma_bf16(v16bf a, v16bf b, v8f c) {
  return __builtin_amdgcn_wmma_f32_16x16x32_bf16(false, a, false, b, (short)0, c,
                                                 false, false);
}

// async 16B global -> LDS copy (per-lane addresses), tracked by ASYNCcnt
__device__ __forceinline__ void async_g2l_b128(uint32_t lds_byte_off,
                                               const void* gaddr) {
  asm volatile("global_load_async_to_lds_b128 %0, %1, off"
               :: "v"(lds_byte_off), "v"((uint64_t)(uintptr_t)gaddr)
               : "memory");
}
__device__ __forceinline__ void wait_async0() {
  asm volatile("s_wait_asynccnt 0x0" ::: "memory");
}
__device__ __forceinline__ uint32_t lds_off(const void* p) {
  return (uint32_t)(uintptr_t)p;  // shared aperture: addr[31:0] = LDS offset
}

__device__ __forceinline__ float hred_max16(float v) {
  v = fmaxf(v, __shfl_xor(v, 1, 32));
  v = fmaxf(v, __shfl_xor(v, 2, 32));
  v = fmaxf(v, __shfl_xor(v, 4, 32));
  v = fmaxf(v, __shfl_xor(v, 8, 32));
  return v;
}
__device__ __forceinline__ float hred_sum16(float v) {
  v += __shfl_xor(v, 1, 32);
  v += __shfl_xor(v, 2, 32);
  v += __shfl_xor(v, 4, 32);
  v += __shfl_xor(v, 8, 32);
  return v;
}

// ---------------------------------------------------------------------------
// f32 -> bf16 cast
// ---------------------------------------------------------------------------
__global__ void cast_f32_bf16(const float* __restrict__ in,
                              __bf16* __restrict__ out, size_t n) {
  size_t i = (size_t)blockIdx.x * blockDim.x + threadIdx.x;
  size_t stride = (size_t)gridDim.x * blockDim.x;
  for (; i < n; i += stride) out[i] = (__bf16)in[i];
}

// ---------------------------------------------------------------------------
// GEMM: out[M,N] = A[M,K](bf16) @ W[N,K](bf16)^T + bias[N]
// block = 256 threads (8 waves, 2(M) x 4(N)), block tile 64(M) x WN*64(N),
// wave tile 32 x WN*16.  K staged 64 at a time via async loads, dbl buffered.
// WN = 2 (N-tile 128) or 4 (N-tile 256; 8 WMMA per 12 ds_load_b128).
// OUT_MODE: 0 = bf16 [M,N]; 1 = bf16 transposed per batch
//           out[(b*N+n)*seq + s], m=b*seq+s; 2 = f32 [M,N]
// ---------------------------------------------------------------------------
#define AT_ELEMS (64 * AROW)

template <int OUT_MODE, int WN>
__global__ __launch_bounds__(256) void gemm_bf16_wmma(
    const __bf16* __restrict__ A, const __bf16* __restrict__ W,
    const float* __restrict__ bias, void* __restrict__ out, int M, int N,
    int K, int seq) {
  constexpr int BT_ROWS = WN * 64;            // B tile rows (= block N)
  constexpr int BT_ELEMS = BT_ROWS * AROW;
  __shared__ __align__(16) __bf16 smem[2 * AT_ELEMS + 2 * BT_ELEMS];

  const int tid = threadIdx.x;
  const int lane = tid & 31;
  const int wid = tid >> 5;
  const int half = lane >> 4;
  const int ln = lane & 15;
  const int mLoc = (wid & 1) * 32;
  const int nLoc = (wid >> 1) * (WN * 16);
  const int mBlk = blockIdx.y * 64;
  const int nBlk = blockIdx.x * (WN * 64);

  auto issue_stage = [&](int buf, int k0) {
    __bf16* at = smem + buf * AT_ELEMS;
    __bf16* bt = smem + 2 * AT_ELEMS + buf * BT_ELEMS;
    // A tile: 64 rows x 64 elems = 512 x 16B chunks; lane-contiguous mapping
#pragma unroll
    for (int u = 0; u < 2; ++u) {
      int c = u * 256 + tid;
      int row = c >> 3, ch = c & 7;
      async_g2l_b128(lds_off(at + row * AROW + ch * 8),
                     A + (size_t)(mBlk + row) * K + k0 + ch * 8);
    }
    // B tile: BT_ROWS x 64 elems = BT_ROWS*8 x 16B chunks
#pragma unroll
    for (int u = 0; u < WN * 2; ++u) {
      int c = u * 256 + tid;
      int row = c >> 3, ch = c & 7;
      async_g2l_b128(lds_off(bt + row * AROW + ch * 8),
                     W + (size_t)(nBlk + row) * K + k0 + ch * 8);
    }
  };

  v8f acc[2][WN] = {};
  const int stages = K / 64;
  issue_stage(0, 0);

  for (int s = 0; s < stages; ++s) {
    wait_async0();        // my loads for buf[cur] have landed in LDS
    __syncthreads();      // everyone's loads landed; prev-stage reads done
    if (s + 1 < stages) issue_stage((s + 1) & 1, (s + 1) * 64);
    const int cur = s & 1;
    const __bf16* at = smem + cur * AT_ELEMS;
    const __bf16* bt = smem + 2 * AT_ELEMS + cur * BT_ELEMS;
#pragma unroll
    for (int kk = 0; kk < 2; ++kk) {
      v16bf af[2], bfr[WN];
#pragma unroll
      for (int i = 0; i < 2; ++i)
        af[i] = load_a_frag(at + (mLoc + i * 16 + ln) * AROW + kk * 32, half);
#pragma unroll
      for (int j = 0; j < WN; ++j)
        bfr[j] = load_b_frag(bt + (nLoc + j * 16 + ln) * AROW + kk * 32 +
                             half * 16);
#pragma unroll
      for (int i = 0; i < 2; ++i)
#pragma unroll
        for (int j = 0; j < WN; ++j)
          acc[i][j] = wmma_bf16(af[i], bfr[j], acc[i][j]);
    }
  }

#pragma unroll
  for (int i = 0; i < 2; ++i) {
#pragma unroll
    for (int j = 0; j < WN; ++j) {
      int n = nBlk + nLoc + j * 16 + ln;
      float bv = bias ? bias[n] : 0.0f;
#pragma unroll
      for (int r = 0; r < 8; ++r) {
        int m = mBlk + mLoc + i * 16 + r + half * 8;
        float v = acc[i][j][r] + bv;
        if (OUT_MODE == 2) {
          ((float*)out)[(size_t)m * N + n] = v;
        } else if (OUT_MODE == 0) {
          ((__bf16*)out)[(size_t)m * N + n] = (__bf16)v;
        } else {
          int b = m / seq, ss = m % seq;
          ((__bf16*)out)[((size_t)b * N + n) * seq + ss] = (__bf16)v;
        }
      }
    }
  }
}

// ---------------------------------------------------------------------------
// Flash MQA attention with async-staged shared K/V tiles.
// grid = (SEQ/128, BS*NH), block = 256 (8 waves, 16 q-rows per wave).
// Q:(b,s,dim) bf16   K:(b,s,128) bf16   Vt:(b,128,seq) bf16
// OT:(b,h,e,t) bf16  (transposed attention output for final GEMM)
// score = -1e-9 where t > q (applied after scale), full-row softmax.
// ---------------------------------------------------------------------------
#define KT_ELEMS (64 * KROW)
#define VT_ELEMS (128 * AROW)

__global__ __launch_bounds__(256) void mqa_flash_attn(
    const __bf16* __restrict__ Q, const __bf16* __restrict__ Kk,
    const __bf16* __restrict__ Vt, __bf16* __restrict__ OT) {
  // K tiles: 2 x 64 x KROW ; V tiles: 2 x 128 x AROW ; P buf: 8 x 16 x AROW
  __shared__ __align__(16) __bf16 smem[2 * KT_ELEMS + 2 * VT_ELEMS +
                                       8 * 16 * AROW];

  const int tid = threadIdx.x;
  const int lane = tid & 31;
  const int wid = tid >> 5;
  const int half = lane >> 4;
  const int ln = lane & 15;
  const int b = blockIdx.y >> 4;
  const int h = blockIdx.y & 15;
  const int qBase = blockIdx.x * 128 + wid * 16;
  const float scale = 0.08838834764831845f;  // 128^-0.5

  auto issue_kv = [&](int buf, int t0) {
    __bf16* kt = smem + buf * KT_ELEMS;
    __bf16* vt = smem + 2 * KT_ELEMS + buf * VT_ELEMS;
    // K tile: 64 rows x 128 elems = 1024 x 16B chunks; lane-contiguous
#pragma unroll
    for (int u = 0; u < 4; ++u) {
      int c = u * 256 + tid;
      int row = c >> 4, ch = c & 15;
      async_g2l_b128(lds_off(kt + row * KROW + ch * 8),
                     Kk + ((size_t)b * SEQ + t0 + row) * HD + ch * 8);
    }
    // V tile: 128 rows x 32 elems = 512 x 16B chunks
#pragma unroll
    for (int u = 0; u < 2; ++u) {
      int c = u * 256 + tid;
      int row = c >> 2, ch = c & 3;
      async_g2l_b128(lds_off(vt + row * AROW + ch * 8),
                     Vt + ((size_t)b * HD + row) * SEQ + t0 + ch * 8);
    }
  };

  // Preload Q A-fragments: 16 rows x 128 e, 4 K-steps of 32.
  v16bf qf[4];
  {
    const __bf16* qrow = Q + ((size_t)b * SEQ + (qBase + ln)) * DIM + h * HD;
#pragma unroll
    for (int kk = 0; kk < 4; ++kk) qf[kk] = load_a_frag(qrow + kk * 32, half);
  }

  float rmax[8], rsum[8];
  v8f oacc[8] = {};
#pragma unroll
  for (int r = 0; r < 8; ++r) { rmax[r] = -3.0e38f; rsum[r] = 0.0f; }

  __bf16* pbuf = smem + 2 * KT_ELEMS + 2 * VT_ELEMS + wid * (16 * AROW);
  const int tiles = SEQ / 64;
  issue_kv(0, 0);

  for (int it = 0; it < tiles; ++it) {
    wait_async0();
    __syncthreads();
    if (it + 1 < tiles) issue_kv((it + 1) & 1, (it + 1) * 64);
    const int cur = it & 1;
    const int t0 = it * 64;
    const __bf16* kt = smem + cur * KT_ELEMS;
    const __bf16* vt = smem + 2 * KT_ELEMS + cur * VT_ELEMS;

    // ---- scores: S(16q x 64t) = Q(16x128) @ K^T from LDS tile ----
    v8f s[4] = {};
#pragma unroll
    for (int kk = 0; kk < 4; ++kk) {
      v16bf kf[4];
#pragma unroll
      for (int nt = 0; nt < 4; ++nt)
        kf[nt] = load_b_frag(kt + (nt * 16 + ln) * KROW + kk * 32 + half * 16);
#pragma unroll
      for (int nt = 0; nt < 4; ++nt) s[nt] = wmma_bf16(qf[kk], kf[nt], s[nt]);
    }

    // ---- mask (analytic tril, fill -1e-9 AFTER scale) + row max ----
    float tmax[8];
#pragma unroll
    for (int r = 0; r < 8; ++r) tmax[r] = -3.0e38f;
#pragma unroll
    for (int nt = 0; nt < 4; ++nt) {
      int t = t0 + nt * 16 + ln;
#pragma unroll
      for (int r = 0; r < 8; ++r) {
        int q = qBase + r + half * 8;
        float v = (t > q) ? -1e-9f : s[nt][r] * scale;
        s[nt][r] = v;
        tmax[r] = fmaxf(tmax[r], v);
      }
    }
#pragma unroll
    for (int r = 0; r < 8; ++r) tmax[r] = hred_max16(tmax[r]);

    // ---- online softmax update ----
    float corr[8];
#pragma unroll
    for (int r = 0; r < 8; ++r) {
      float mn = fmaxf(rmax[r], tmax[r]);
      corr[r] = __expf(rmax[r] - mn);
      rmax[r] = mn;
    }
    float tsum[8];
#pragma unroll
    for (int r = 0; r < 8; ++r) tsum[r] = 0.0f;
#pragma unroll
    for (int nt = 0; nt < 4; ++nt) {
#pragma unroll
      for (int r = 0; r < 8; ++r) {
        float p = __expf(s[nt][r] - rmax[r]);
        s[nt][r] = p;
        tsum[r] += p;
      }
    }
#pragma unroll
    for (int r = 0; r < 8; ++r) {
      tsum[r] = hred_sum16(tsum[r]);
      rsum[r] = rsum[r] * corr[r] + tsum[r];
    }
#pragma unroll
    for (int et = 0; et < 8; ++et)
#pragma unroll
      for (int r = 0; r < 8; ++r) oacc[et][r] *= corr[r];

    // ---- P (D-fragment layout) -> per-wave padded LDS tile ----
#pragma unroll
    for (int nt = 0; nt < 4; ++nt) {
      int col = nt * 16 + ln;
#pragma unroll
      for (int r = 0; r < 8; ++r)
        pbuf[(r + half * 8) * AROW + col] = (__bf16)s[nt][r];
    }
    asm volatile("s_wait_dscnt 0" ::: "memory");  // per-wave LDS RAW fence

    // ---- O += P(16x64) @ V(64x128) from LDS V tile ----
#pragma unroll
    for (int kk = 0; kk < 2; ++kk) {
      v16bf pf = load_a_frag(pbuf + ln * AROW + kk * 32, half);
#pragma unroll
      for (int et = 0; et < 8; ++et) {
        v16bf vf = load_b_frag(vt + (et * 16 + ln) * AROW + kk * 32 + half * 16);
        oacc[et] = wmma_bf16(pf, vf, oacc[et]);
      }
    }
    asm volatile("s_wait_dscnt 0" ::: "memory");  // WAR before next P stores
  }

  // ---- finalize: O / rowsum, store transposed bf16 OT[b][h][e][q] ----
  float inv[8];
#pragma unroll
  for (int r = 0; r < 8; ++r) inv[r] = 1.0f / rsum[r];
#pragma unroll
  for (int et = 0; et < 8; ++et) {
    int e = et * 16 + ln;
    __bf16* dst = OT + (((size_t)b * NH + h) * HD + e) * SEQ;
#pragma unroll
    for (int r = 0; r < 8; ++r) {
      int q = qBase + r + half * 8;
      dst[q] = (__bf16)(oacc[et][r] * inv[r]);
    }
  }
}

// ---------------------------------------------------------------------------
// host launcher
// ---------------------------------------------------------------------------
extern "C" void kernel_launch(void* const* d_in, const int* in_sizes, int n_in,
                              void* d_out, int out_size, void* d_ws,
                              size_t ws_size, hipStream_t stream) {
  (void)in_sizes; (void)n_in; (void)out_size; (void)ws_size;
  const float* x   = (const float*)d_in[0];
  // d_in[1] = masks: tril ones, applied analytically — not read.
  const float* q_w = (const float*)d_in[2];
  const float* q_b = (const float*)d_in[3];
  const float* k_w = (const float*)d_in[4];
  const float* k_b = (const float*)d_in[5];
  const float* v_w = (const float*)d_in[6];
  const float* v_b = (const float*)d_in[7];
  const float* o_w = (const float*)d_in[8];
  const float* o_b = (const float*)d_in[9];
  float* out = (float*)d_out;

  char* ws = (char*)d_ws;
  size_t off = 0;
  auto carve = [&](size_t bytes) { char* p = ws + off; off += bytes; return p; };
  __bf16* xb    = (__bf16*)carve((size_t)MTOT * DIM * 2);  // 33.5 MB
  __bf16* qwb   = (__bf16*)carve((size_t)DIM * DIM * 2);   //  8.4 MB
  __bf16* kwb   = (__bf16*)carve((size_t)HD * DIM * 2);    //  0.5 MB
  __bf16* vwb   = (__bf16*)carve((size_t)HD * DIM * 2);    //  0.5 MB
  __bf16* owb   = (__bf16*)carve((size_t)DIM * DIM * 2);   //  8.4 MB
  __bf16* qb    = (__bf16*)carve((size_t)MTOT * DIM * 2);  // 33.5 MB
  __bf16* kb    = (__bf16*)carve((size_t)MTOT * HD * 2);   //  2.1 MB
  __bf16* vtb   = (__bf16*)carve((size_t)MTOT * HD * 2);   //  2.1 MB (transposed)
  __bf16* attnT = (__bf16*)carve((size_t)MTOT * DIM * 2);  // 33.5 MB

  // 1) casts to bf16
  cast_f32_bf16<<<1024, 256, 0, stream>>>(x,   xb,  (size_t)MTOT * DIM);
  cast_f32_bf16<<<512,  256, 0, stream>>>(q_w, qwb, (size_t)DIM * DIM);
  cast_f32_bf16<<<128,  256, 0, stream>>>(k_w, kwb, (size_t)HD * DIM);
  cast_f32_bf16<<<128,  256, 0, stream>>>(v_w, vwb, (size_t)HD * DIM);
  cast_f32_bf16<<<512,  256, 0, stream>>>(o_w, owb, (size_t)DIM * DIM);

  // 2) projections  (WN=4 -> 64x256 block tile for the big-N GEMM)
  gemm_bf16_wmma<0, 4><<<dim3(DIM / 256, MTOT / 64), 256, 0, stream>>>(
      xb, qwb, q_b, qb, MTOT, DIM, DIM, SEQ);               // Q: 8192x2048
  gemm_bf16_wmma<0, 2><<<dim3(1, MTOT / 64), 256, 0, stream>>>(
      xb, kwb, k_b, kb, MTOT, HD, DIM, SEQ);                // K: 8192x128
  gemm_bf16_wmma<1, 2><<<dim3(1, MTOT / 64), 256, 0, stream>>>(
      xb, vwb, v_b, vtb, MTOT, HD, DIM, SEQ);               // V^T: (b,128,seq)

  // 3) flash attention -> transposed output
  mqa_flash_attn<<<dim3(SEQ / 128, BS * NH), 256, 0, stream>>>(qb, kb, vtb,
                                                               attnT);

  // 4) output projection (f32 result)
  gemm_bf16_wmma<2, 4><<<dim3(DIM / 256, MTOT / 64), 256, 0, stream>>>(
      attnT, owb, o_b, out, MTOT, DIM, DIM, SEQ);
}
